// ConvEncoder2_63788854280432
// MI455X (gfx1250) — matrix operand order repaired and checked by reading, not compiled, not run on hardware
//
#include <hip/hip_runtime.h>
#include <stdint.h>

// ---------------------------------------------------------------------------
// CDNA5 (gfx1250) wave32 WMMA types
// ---------------------------------------------------------------------------
typedef __attribute__((ext_vector_type(16))) __bf16 v16bf;
typedef __attribute__((ext_vector_type(8)))  float  v8f;
typedef __attribute__((ext_vector_type(4)))  unsigned int u32x4;
typedef __attribute__((ext_vector_type(4)))  int i32x4;

#define NPTS  65536   // B*T
#define TLOG  15      // T = 32768

#if __has_builtin(__builtin_amdgcn_global_load_async_to_lds_b128)
#define HAVE_ASYNC_LDS 1
#else
#define HAVE_ASYNC_LDS 0
#endif

__device__ __forceinline__ unsigned short f2bfbits(float f) {
  union { float f; unsigned int u; } v; v.f = f;
  unsigned int r = v.u + 0x7FFFu + ((v.u >> 16) & 1u);
  return (unsigned short)(r >> 16);
}
__device__ __forceinline__ float bfbits2f(unsigned short h) {
  union { unsigned int u; float f; } v; v.u = ((unsigned int)h) << 16;
  return v.f;
}

union AFrag { v16bf v; unsigned short s[16]; u32x4 q[2]; };

// --- async global->LDS copy of one 16B chunk (per-lane addresses) ----------
// builtin signature (from hipcc diagnostic): (v4i AS1*, v4i AS3*, Imm, Imm)
__device__ __forceinline__ void async_copy16(const unsigned short* g, unsigned short* l) {
#if HAVE_ASYNC_LDS
  __builtin_amdgcn_global_load_async_to_lds_b128(
      (__attribute__((address_space(1))) i32x4*)(i32x4*)(unsigned short*)g,
      (__attribute__((address_space(3))) i32x4*)(i32x4*)l, 0, 0);
#else
  *(u32x4*)l = *(const u32x4*)g;
#endif
}
// each wave has 4 async-copy instructions per panel in flight; waiting to <=4
// drains the previous panel while the next streams in (in-order completion)
__device__ __forceinline__ void async_wait4() {
#if HAVE_ASYNC_LDS
#if __has_builtin(__builtin_amdgcn_s_wait_asynccnt)
  __builtin_amdgcn_s_wait_asynccnt(4);
#else
  asm volatile("s_wait_asynccnt 0x4" ::: "memory");
#endif
#endif
}

// LDS B-panel: 64 cols x 64 K of bf16, row stride padded to 72 ushorts
// (144 B: 16B-aligned for b128, spreads banks). Double buffered: 18 KB.
#define BROW 72

// ---------------------------------------------------------------------------
// Generic dense GEMM: each wave computes a 16-row x 64-col strip, K in
// chunks of 64 (2 WMMA K-steps per chunk -> 8 WMMAs per barrier pair).
// A is a "dual source" view: k in [0,KA) from xa (per-row), k in [KA,KA+KB)
// from xb (per-row, or per-batch broadcast when xb_bcast). Weights Wt are
// pre-transposed bf16 (N x Ktot); the 64x64 weight panel is staged through
// LDS with double-buffered, branch-free async copies.
// ---------------------------------------------------------------------------
__global__ __launch_bounds__(128) void gemm_lin_k(
    const unsigned short* __restrict__ xa, int lda,
    const unsigned short* __restrict__ xb, int ldb, int xb_bcast,
    int KA, int KB, int relu_in,
    const unsigned short* __restrict__ Wt,
    const float* __restrict__ bias,
    const float* __restrict__ addsrc,
    unsigned short* __restrict__ outb, int relu_out,
    float* __restrict__ outf, int N)
{
  __shared__ unsigned short bsm[2][64 * BROW];

  const int tid  = threadIdx.x;
  const int lane = tid & 31;
  const int wave = tid >> 5;
  const int half = lane >> 4;
  const int l15  = lane & 15;
  const int row0 = (blockIdx.x * 4 + wave) * 16;
  const int col0 = blockIdx.y * 64;
  const int Ktot = KA + KB;
  const int r = row0 + l15;

  // issue the 64x64 weight panel for k-chunk k0 into buffer `buf`
  auto copyB = [&](int buf, int k0) {
#pragma unroll
    for (int i = 0; i < 4; ++i) {
      const int q  = tid + i * 128;   // 512 chunks of 16B
      const int cc = q >> 3;          // col within panel
      const int ch = q & 7;           // 16B chunk within the 64-K row
      async_copy16(Wt + (size_t)(col0 + cc) * Ktot + k0 + ch * 8,
                   &bsm[buf][cc * BROW + ch * 8]);
    }
  };

  v8f acc[4];
  for (int j = 0; j < 4; ++j)
    for (int e = 0; e < 8; ++e) acc[j][e] = 0.0f;

  copyB(0, 0);

  for (int k0 = 0; k0 < Ktot; k0 += 64) {
    const int cur   = (k0 >> 6) & 1;
    const int knext = (k0 + 64 < Ktot) ? (k0 + 64) : k0;  // dummy re-copy at end
    copyB(cur ^ 1, knext);

    // two A fragments from global (overlap the async weight copy);
    // the [k0,k0+64) chunk lies entirely in one source (KA multiple of 64)
    const unsigned short* pb;
    if (k0 < KA) {
      pb = xa + (size_t)r * lda + k0;
    } else {
      const int rr = xb_bcast ? (r >> TLOG) : r;
      pb = xb + (size_t)rr * ldb + (k0 - KA);
    }
    AFrag A0, A1;
    {
      const unsigned short* p0 = pb + half * 8;
      A0.q[0] = *(const u32x4*)(p0);
      A0.q[1] = *(const u32x4*)(p0 + 16);
      const unsigned short* p1 = pb + 32 + half * 8;
      A1.q[0] = *(const u32x4*)(p1);
      A1.q[1] = *(const u32x4*)(p1 + 16);
    }
    if (relu_in) {
#pragma unroll
      for (int i = 0; i < 16; ++i) {
        A0.s[i] = (A0.s[i] & 0x8000u) ? (unsigned short)0 : A0.s[i];
        A1.s[i] = (A1.s[i] & 0x8000u) ? (unsigned short)0 : A1.s[i];
      }
    }

    async_wait4();
    __syncthreads();

    // load all B fragments first, then issue all WMMAs
    AFrag B0[4], B1[4];
#pragma unroll
    for (int j = 0; j < 4; ++j) {
      const unsigned short* p = &bsm[cur][(j * 16 + l15) * BROW + half * 8];
      B0[j].q[0] = *(const u32x4*)(p);
      B0[j].q[1] = *(const u32x4*)(p + 16);
      B1[j].q[0] = *(const u32x4*)(p + 32);
      B1[j].q[1] = *(const u32x4*)(p + 48);
    }
#pragma unroll
    for (int j = 0; j < 4; ++j)
      acc[j] = __builtin_amdgcn_wmma_f32_16x16x32_bf16(
          false, A0.v, false, B0[j].v, (short)0, acc[j], false, false);
#pragma unroll
    for (int j = 0; j < 4; ++j)
      acc[j] = __builtin_amdgcn_wmma_f32_16x16x32_bf16(
          false, A1.v, false, B1[j].v, (short)0, acc[j], false, false);

    __syncthreads();   // panel fully consumed before it is overwritten
  }

  const int rbase = row0 + half * 8;
#pragma unroll
  for (int j = 0; j < 4; ++j) {
    const int c = col0 + j * 16 + l15;
    const float bv = bias ? bias[c] : 0.0f;
#pragma unroll
    for (int e = 0; e < 8; ++e) {
      const int rr = rbase + e;
      float v = acc[j][e] + bv;
      if (addsrc) v += addsrc[(size_t)rr * N + c];
      if (outf) outf[(size_t)rr * N + c] = v;
      if (outb) {
        float vv = (relu_out && v < 0.0f) ? 0.0f : v;
        outb[(size_t)rr * N + c] = f2bfbits(vv);
      }
    }
  }
}

// ---------------------------------------------------------------------------
// Implicit-GEMM conv (NCHW, OIHW weights flat as N x (Cin*KS*KS) bf16).
// Each wave: 16 output pixels x 64 output channels, K in chunks of 64;
// weight panel staged in LDS (double buffered async), A gathered from
// global with padding checks.
// ---------------------------------------------------------------------------
template <int INF32, int OUTF32, int KS>
__global__ __launch_bounds__(128) void conv_wmma_k(
    const void* __restrict__ in_,
    const unsigned short* __restrict__ Wt,
    const float* __restrict__ bias,
    void* __restrict__ out_,
    int Cin, int H, int W, int OC, int OH, int OW,
    int stride, int pad, int relu_out)
{
  __shared__ unsigned short bsm[2][64 * BROW];

  const int tid  = threadIdx.x;
  const int lane = tid & 31;
  const int wave = tid >> 5;
  const int half = lane >> 4;
  const int l15  = lane & 15;
  const int b    = blockIdx.z;
  const int m0   = (blockIdx.x * 4 + wave) * 16;
  const int col0 = blockIdx.y * 64;
  const int Kg   = Cin * KS * KS;
  const int m  = m0 + l15;
  const int oy = m / OW, ox = m - oy * OW;
  const int iy0 = oy * stride - pad, ix0 = ox * stride - pad;
  const float*          inf = (const float*)in_ + (size_t)b * Cin * H * W;
  const unsigned short* inb = (const unsigned short*)in_ + (size_t)b * Cin * H * W;

  auto copyB = [&](int buf, int k0) {
#pragma unroll
    for (int i = 0; i < 4; ++i) {
      const int q  = tid + i * 128;
      const int cc = q >> 3;
      const int ch = q & 7;
      async_copy16(Wt + (size_t)(col0 + cc) * Kg + k0 + ch * 8,
                   &bsm[buf][cc * BROW + ch * 8]);
    }
  };

  // gather one 16x32 im2col A fragment with zero padding
  auto gatherA = [&](int kbase) {
    AFrag A;
#pragma unroll
    for (int e = 0; e < 16; ++e) {
      const int kk = (e < 8) ? (half * 8 + e) : (16 + half * 8 + (e - 8));
      const int kg = kbase + kk;
      const int ic = kg / (KS * KS);
      const int rk = kg - ic * (KS * KS);
      const int ky = rk / KS;
      const int kx = rk - ky * KS;
      const int iy = iy0 + ky, ix = ix0 + kx;
      unsigned short vb = 0;
      if ((unsigned)iy < (unsigned)H && (unsigned)ix < (unsigned)W) {
        const size_t idx = ((size_t)ic * H + iy) * W + ix;
        vb = INF32 ? f2bfbits(inf[idx]) : inb[idx];
      }
      A.s[e] = vb;
    }
    return A;
  };

  v8f acc[4];
  for (int j = 0; j < 4; ++j)
    for (int e = 0; e < 8; ++e) acc[j][e] = 0.0f;

  copyB(0, 0);

  for (int k0 = 0; k0 < Kg; k0 += 64) {
    const int cur   = (k0 >> 6) & 1;
    const int knext = (k0 + 64 < Kg) ? (k0 + 64) : k0;   // dummy re-copy at end
    copyB(cur ^ 1, knext);

    AFrag A0 = gatherA(k0);
    AFrag A1 = gatherA(k0 + 32);

    async_wait4();
    __syncthreads();

    AFrag B0[4], B1[4];
#pragma unroll
    for (int j = 0; j < 4; ++j) {
      const unsigned short* p = &bsm[cur][(j * 16 + l15) * BROW + half * 8];
      B0[j].q[0] = *(const u32x4*)(p);
      B0[j].q[1] = *(const u32x4*)(p + 16);
      B1[j].q[0] = *(const u32x4*)(p + 32);
      B1[j].q[1] = *(const u32x4*)(p + 48);
    }
#pragma unroll
    for (int j = 0; j < 4; ++j)
      acc[j] = __builtin_amdgcn_wmma_f32_16x16x32_bf16(
          false, A0.v, false, B0[j].v, (short)0, acc[j], false, false);
#pragma unroll
    for (int j = 0; j < 4; ++j)
      acc[j] = __builtin_amdgcn_wmma_f32_16x16x32_bf16(
          false, A1.v, false, B1[j].v, (short)0, acc[j], false, false);

    __syncthreads();
  }

  const int mbase = m0 + half * 8;
  const size_t ohow = (size_t)OH * OW;
#pragma unroll
  for (int j = 0; j < 4; ++j) {
    const int oc = col0 + j * 16 + l15;
    const float bv = bias[oc];
#pragma unroll
    for (int e = 0; e < 8; ++e) {
      const int mm = mbase + e;
      float v = acc[j][e] + bv;
      if (relu_out && v < 0.0f) v = 0.0f;
      const size_t oidx = ((size_t)b * OC + oc) * ohow + mm;
      if (OUTF32) ((float*)out_)[oidx] = v;
      else        ((unsigned short*)out_)[oidx] = f2bfbits(v);
    }
  }
}

// ---------------------------------------------------------------------------
// Small helper kernels
// ---------------------------------------------------------------------------
__global__ void fc_pos_k(const float* __restrict__ in, const float* __restrict__ w,
                         const float* __restrict__ bias, unsigned short* __restrict__ out)
{
  const int i = blockIdx.x * blockDim.x + threadIdx.x;   // over NPTS*256
  const int row = i >> 8;
  const int col = i & 255;
  const float* q = in + (size_t)row * 3;
  float v = bias[col] + q[0] * w[col] + q[1] * w[256 + col] + q[2] * w[512 + col];
  out[i] = f2bfbits(v);
}

__global__ void pool_max_k(const unsigned short* __restrict__ net,
                           unsigned short* __restrict__ pooled)
{
  const int bc = blockIdx.x;                 // b*128 + ch
  const int b = bc >> 7, ch = bc & 127;
  const unsigned short* p = net + ((size_t)b << TLOG) * 128 + ch;
  float m = -3.402823e38f;
  for (int t = threadIdx.x; t < (1 << TLOG); t += 256)
    m = fmaxf(m, bfbits2f(p[(size_t)t * 128]));
  __shared__ float red[256];
  red[threadIdx.x] = m;
  __syncthreads();
  for (int s = 128; s > 0; s >>= 1) {
    if (threadIdx.x < s) red[threadIdx.x] = fmaxf(red[threadIdx.x], red[threadIdx.x + s]);
    __syncthreads();
  }
  if (threadIdx.x == 0) pooled[bc] = f2bfbits(red[0]);
}

__global__ void scatter_k(const float* __restrict__ pts, const float* __restrict__ c,
                          float* __restrict__ plane, float* __restrict__ cnt)
{
  const int pt = blockIdx.x;                 // point index (B*T)
  const int b  = pt >> TLOG;
  const float s = 1.0f / (1.0f + 0.1f + 1e-3f);
  float xx = pts[(size_t)pt * 3 + 0] * s + 0.5f;
  float zz = pts[(size_t)pt * 3 + 2] * s + 0.5f;
  xx = fminf(fmaxf(xx, 0.0f), 1.0f - 1e-3f);
  zz = fminf(fmaxf(zz, 0.0f), 1.0f - 1e-3f);
  const int cell = (int)(xx * 512.0f) + 512 * (int)(zz * 512.0f);
  const int ch = threadIdx.x;                // 128 channels
  atomicAdd(plane + (((size_t)b * 128 + ch) * 262144 + cell), c[(size_t)pt * 128 + ch]);
  if (ch == 0) atomicAdd(cnt + (size_t)b * 262144 + cell, 1.0f);
}

__global__ void divide_k(float* __restrict__ plane, const float* __restrict__ cnt)
{
  const size_t i = (size_t)blockIdx.x * 256 + threadIdx.x;  // over 2*128*262144
  const size_t cell = i & 262143;
  const size_t b = i >> 25;                                 // / (128*262144)
  plane[i] *= 1.0f / fmaxf(cnt[(b << 18) + cell], 1.0f);
}

__global__ void transpose_w_k(const float* __restrict__ w, unsigned short* __restrict__ wt,
                              int K, int N)
{
  const int i = blockIdx.x * blockDim.x + threadIdx.x;
  if (i >= K * N) return;
  const int k = i / N, n = i - k * N;
  wt[(size_t)n * K + k] = f2bfbits(w[i]);   // (K x N) f32 -> (N x K) bf16
}

__global__ void cast_w_k(const float* __restrict__ w, unsigned short* __restrict__ wt, int n)
{
  const int i = blockIdx.x * blockDim.x + threadIdx.x;
  if (i < n) wt[i] = f2bfbits(w[i]);
}

// ---------------------------------------------------------------------------
// Host launcher
// ---------------------------------------------------------------------------
extern "C" void kernel_launch(void* const* d_in, const int* in_sizes, int n_in,
                              void* d_out, int out_size, void* d_ws, size_t ws_size,
                              hipStream_t stream)
{
  (void)in_sizes; (void)n_in; (void)out_size; (void)ws_size;

  // d_in order: p_occ, occ, inputs, then params flattened (jax pytree: sorted
  // dict keys): blocks[0..2]{fc0_b,fc0_w,fc1_b,fc1_w,sc_w}, convs[0..3]{b,w},
  // fc_pos_b, fc_pos_w, fc_z_b, fc_z_w, logstd_b, logstd_w, mean_b, mean_w
  const float* inputs = (const float*)d_in[2];
  int p = 3;
  const float *fc0_b[3], *fc0_w[3], *fc1_b[3], *fc1_w[3], *sc_w[3];
  for (int i = 0; i < 3; ++i) {
    fc0_b[i] = (const float*)d_in[p++];
    fc0_w[i] = (const float*)d_in[p++];
    fc1_b[i] = (const float*)d_in[p++];
    fc1_w[i] = (const float*)d_in[p++];
    sc_w[i]  = (const float*)d_in[p++];
  }
  const float *conv_b[4], *conv_w[4];
  for (int i = 0; i < 4; ++i) {
    conv_b[i] = (const float*)d_in[p++];
    conv_w[i] = (const float*)d_in[p++];
  }
  const float* fc_pos_b = (const float*)d_in[p++];
  const float* fc_pos_w = (const float*)d_in[p++];
  const float* fc_z_b   = (const float*)d_in[p++];
  const float* fc_z_w   = (const float*)d_in[p++];
  const float* logstd_b = (const float*)d_in[p++];
  const float* logstd_w = (const float*)d_in[p++];
  const float* mean_b   = (const float*)d_in[p++];
  const float* mean_w   = (const float*)d_in[p++];

  // Workspace layout (buffers reused across pipeline phases)
  const size_t MB = 1024u * 1024u;
  char* ws = (char*)d_ws;
  float*          plane  = (float*)(ws);                     // 256 MB f32
  float*          cnt    = (float*)(ws + 256 * MB);          //   2 MB f32
  unsigned short* wa     = (unsigned short*)(ws + 258 * MB); //  72 MB bf16 weights
  char*           bufA   = ws + 330 * MB;                    //  64 MB
  char*           bufB   = ws + 394 * MB;                    //  32 MB
  char*           bufC   = ws + 426 * MB;                    //  34 MB
  unsigned short* pooled = (unsigned short*)(ws + 460 * MB); //  tiny

  unsigned short* net256 = (unsigned short*)bufA;            // MLP phase
  unsigned short* net128 = (unsigned short*)bufB;
  unsigned short* hrelu  = (unsigned short*)(bufB + 16 * MB);
  float*          tmpf   = (float*)bufC;                     // shortcut / c features

  // bf16 weight arena offsets (elements)
  size_t wo = 0;
  size_t oW0[3], oW1[3], oWs[3];
  for (int i = 0; i < 3; ++i) {
    oW0[i] = wo; wo += 32768;
    oW1[i] = wo; wo += 16384;
    oWs[i] = wo; wo += 32768;
  }
  size_t oWz = wo; wo += 32768;
  size_t oC[4];
  const int csz[4] = {294912, 1179648, 4718592, 18874368};
  for (int i = 0; i < 4; ++i) { oC[i] = wo; wo += (size_t)csz[i]; }
  size_t oM = wo; wo += 4194304;
  size_t oL = wo; wo += 4194304;

  // zero plane + counts (scatter accumulates)
  (void)hipMemsetAsync(ws, 0, 258 * MB, stream);

  // weight conversion: linear weights transposed to (N x K), conv weights flat
  for (int i = 0; i < 3; ++i) {
    transpose_w_k<<<dim3((256 * 128 + 255) / 256), 256, 0, stream>>>(fc0_w[i], wa + oW0[i], 256, 128);
    transpose_w_k<<<dim3((128 * 128 + 255) / 256), 256, 0, stream>>>(fc1_w[i], wa + oW1[i], 128, 128);
    transpose_w_k<<<dim3((256 * 128 + 255) / 256), 256, 0, stream>>>(sc_w[i],  wa + oWs[i], 256, 128);
  }
  transpose_w_k<<<dim3((256 * 128 + 255) / 256), 256, 0, stream>>>(fc_z_w, wa + oWz, 256, 128);
  for (int i = 0; i < 4; ++i)
    cast_w_k<<<dim3((csz[i] + 255) / 256), 256, 0, stream>>>(conv_w[i], wa + oC[i], csz[i]);
  cast_w_k<<<dim3((4194304 + 255) / 256), 256, 0, stream>>>(mean_w,   wa + oM, 4194304);
  cast_w_k<<<dim3((4194304 + 255) / 256), 256, 0, stream>>>(logstd_w, wa + oL, 4194304);

  // fc_pos: (B*T,3) @ (3,256) + b -> net256 (bf16)
  fc_pos_k<<<dim3(NPTS), 256, 0, stream>>>(inputs, fc_pos_w, fc_pos_b, net256);

  // ResNet blocks (concat folded into dual-source A view)
  const dim3 gg(NPTS / 64, 2, 1);  // (M/64, N/64) with N=128
  for (int i = 0; i < 3; ++i) {
    const unsigned short* xa = (i == 0) ? net256 : net128;
    const int lda            = (i == 0) ? 256 : 128;
    const unsigned short* xb = (i == 0) ? (net256 + 128) : pooled;
    const int ldb            = (i == 0) ? 256 : 128;
    const int bcast          = (i == 0) ? 0 : 1;
    // h = relu(fc0(relu(x)) + b0)
    gemm_lin_k<<<gg, 128, 0, stream>>>(xa, lda, xb, ldb, bcast, 128, 128, 1,
                                       wa + oW0[i], fc0_b[i], nullptr,
                                       hrelu, 1, nullptr, 128);
    // tmp = x @ sc_w  (raw x, no bias)
    gemm_lin_k<<<gg, 128, 0, stream>>>(xa, lda, xb, ldb, bcast, 128, 128, 0,
                                       wa + oWs[i], nullptr, nullptr,
                                       nullptr, 0, tmpf, 128);
    // net128 = fc1(h) + b1 + tmp
    gemm_lin_k<<<gg, 128, 0, stream>>>(hrelu, 128, nullptr, 0, 0, 128, 0, 0,
                                       wa + oW1[i], fc1_b[i], tmpf,
                                       net128, 0, nullptr, 128);
    // pooled = max over T
    pool_max_k<<<dim3(256), 256, 0, stream>>>(net128, pooled);
  }

  // fc_z: c = concat(net128, pooled) @ Wz + bz  (f32 out)
  gemm_lin_k<<<gg, 128, 0, stream>>>(net128, 128, pooled, 128, 1, 128, 128, 0,
                                     wa + oWz, fc_z_b, nullptr,
                                     nullptr, 0, tmpf, 128);

  // scatter-mean into the 512x512x128 plane (NCHW)
  scatter_k<<<dim3(NPTS), 128, 0, stream>>>(inputs, tmpf, plane, cnt);
  divide_k<<<dim3(262144), 256, 0, stream>>>(plane, cnt);

  // conv stack (implicit GEMM, bf16 WMMA, relu folded into stores)
  conv_wmma_k<1, 0, 3><<<dim3(1024, 4, 2), 128, 0, stream>>>(
      plane, wa + oC[0], conv_b[0], bufA, 128, 512, 512, 256, 256, 256, 2, 1, 1);
  conv_wmma_k<0, 0, 3><<<dim3(256, 8, 2), 128, 0, stream>>>(
      bufA, wa + oC[1], conv_b[1], bufB, 256, 256, 256, 512, 128, 128, 2, 1, 1);
  conv_wmma_k<0, 0, 3><<<dim3(64, 16, 2), 128, 0, stream>>>(
      bufB, wa + oC[2], conv_b[2], bufC, 512, 128, 128, 1024, 64, 64, 2, 1, 1);
  conv_wmma_k<0, 0, 3><<<dim3(16, 32, 2), 128, 0, stream>>>(
      bufC, wa + oC[3], conv_b[3], bufA, 1024, 64, 64, 2048, 32, 32, 2, 1, 0);

  // 1x1 heads -> f32 d_out: mean then log_std
  conv_wmma_k<0, 1, 1><<<dim3(16, 32, 2), 128, 0, stream>>>(
      bufA, wa + oM, mean_b, d_out, 2048, 32, 32, 2048, 32, 32, 1, 0, 0);
  conv_wmma_k<0, 1, 1><<<dim3(16, 32, 2), 128, 0, stream>>>(
      bufA, wa + oL, logstd_b, (float*)d_out + 4194304, 2048, 32, 32, 2048, 32, 32, 1, 0, 0);
}
